// PathAgg_att_sample_layer_12558484373609
// MI455X (gfx1250) — matrix operand order repaired and checked by reading, not compiled, not run on hardware
//
#include <hip/hip_runtime.h>
#include <stdint.h>

#define N_NODES 50000
#define P_PATHS 200000
#define LPATH   6
#define IN_DIM  128
#define OUT_DIM 128
#define HEADS   4
#define WAVES   8      // waves per block in the GRU kernel

// LDS layout (dynamic shared memory). Rows padded 256B -> 272B (stride 68 dwords,
// 68 % 64 = 4 banks) so 16 lanes reading 16 consecutive rows are conflict-free.
#define W_ROW_B   272
#define LDS_W1    ((size_t)384 * W_ROW_B)          // one weight matrix, padded
#define LDS_B_OFF (2 * LDS_W1)                     // combined-bias table
#define LDS_B_SZ  (128 * 16)                       // 128 dims x float4 {b_r,b_z,b_in,b_hn}
#define LDS_H_OFF (LDS_B_OFF + LDS_B_SZ)
#define H_ROW_B   272
#define LDS_H_PW  ((size_t)16 * H_ROW_B)           // per-wave h staging
#define SMEM_BYTES (LDS_H_OFF + WAVES * LDS_H_PW)  // ~240 KB < 320 KB

typedef __attribute__((ext_vector_type(16))) __bf16 v16bf;
typedef __attribute__((ext_vector_type(8)))  __bf16 v8bf;
typedef __attribute__((ext_vector_type(8)))  float  v8f;

static __device__ __forceinline__ uint16_t f2bf(float f) {
  union { float f; uint32_t u; } c; c.f = f;
  uint32_t u = c.u;
  uint32_t r = (u + 0x7FFFu + ((u >> 16) & 1u)) >> 16;   // round-to-nearest-even
  return (uint16_t)r;
}
static __device__ __forceinline__ float bf2f(uint16_t b) {
  union { uint32_t u; float f; } c; c.u = ((uint32_t)b) << 16;
  return c.f;
}
static __device__ __forceinline__ float sigm(float x) {
  return 1.0f / (1.0f + __expf(-x));
}

// ---------------- utility kernels ----------------

__global__ void k_cvt_bf16(const float* __restrict__ src, uint16_t* __restrict__ dst, int n) {
  int i = blockIdx.x * blockDim.x + threadIdx.x;
  if (i < n) dst[i] = f2bf(src[i]);
}

__global__ void k_zero(float* __restrict__ p, long long n) {
  long long i = (long long)blockIdx.x * blockDim.x + threadIdx.x;
  long long stride = (long long)gridDim.x * blockDim.x;
  for (; i < n; i += stride) p[i] = 0.0f;
}

// ---------------- phase 1: GRU over paths + unnormalized attention ----------------
__global__ __launch_bounds__(32 * WAVES)
void k_gru_paths(const uint16_t* __restrict__ xbf,      // N x 128 bf16
                 const long long* __restrict__ path,    // P x 6 int64
                 const uint16_t* __restrict__ wihbf,    // 384 x 128 bf16 row-major
                 const uint16_t* __restrict__ whhbf,    // 384 x 128 bf16 row-major
                 const float* __restrict__ bih,         // 384
                 const float* __restrict__ bhh,         // 384
                 const float* __restrict__ amat,        // 128 x 4
                 float* __restrict__ path_emb,          // P x 128
                 float* __restrict__ att_un,            // P x 4
                 float* __restrict__ att_sum)           // N x 4 (atomic accumulate)
{
  extern __shared__ char smem[];

  const int tid   = threadIdx.x;
  const int lane  = tid & 31;
  const int wv    = tid >> 5;
  const int p0    = (blockIdx.x * WAVES + wv) * 16;

  // ---- cooperative staging into LDS (all waves participate) ----
  {
    #pragma unroll 1
    for (int c = tid; c < 2 * 384 * 8; c += 32 * WAVES) {
      const int mat = c >> 12;          // 3072 16B-chunks per matrix
      const int rc  = c & 3071;
      const int row = rc >> 3;
      const int j   = rc & 7;
      const uint16_t* src = (mat ? whhbf : wihbf) + row * 128 + j * 8;
      uint4 v = *(const uint4*)src;
      *(uint4*)(smem + (size_t)mat * LDS_W1 + (size_t)row * W_ROW_B + j * 16) = v;
    }
    // combined bias table: per dim d -> {bih_r+bhh_r, bih_z+bhh_z, bih_n, bhh_n}
    #pragma unroll 1
    for (int c = tid; c < 512; c += 32 * WAVES) {
      const int d     = c >> 2;
      const int which = c & 3;
      float v;
      if      (which == 0) v = bih[d]       + bhh[d];
      else if (which == 1) v = bih[128 + d] + bhh[128 + d];
      else if (which == 2) v = bih[256 + d];
      else                 v = bhh[256 + d];
      *(float*)(smem + LDS_B_OFF + (size_t)c * 4) = v;
    }
  }
  __syncthreads();

  if (p0 >= P_PATHS) return;           // wave-uniform: EXEC stays all-ones

  char* hbuf = smem + LDS_H_OFF + (size_t)wv * LDS_H_PW;
  const int col   = lane & 15;         // tile column / path row for A loads
  const int khalf = lane >> 4;         // which K half this lane carries
  const int m     = col;

  // h0 = 0 : zero the per-wave staging buffer (4352 B = 272 x16B chunks)
  {
    uint4 z; z.x = z.y = z.z = z.w = 0u;
    #pragma unroll
    for (int i = lane; i < 272; i += 32) *(uint4*)(hbuf + (size_t)i * 16) = z;
  }

  // persistent hidden state in C/D layout: h[dt][e] = h(path p0+e+8*khalf, dim dt*16+col)
  v8f h[8];
  #pragma unroll
  for (int dt = 0; dt < 8; ++dt) {
    v8f z = {0.f,0.f,0.f,0.f,0.f,0.f,0.f,0.f};
    h[dt] = z;
  }

  #pragma unroll 1
  for (int t = 0; t < LPATH; ++t) {
    // launder a zero so LDS weight/bias addresses are loop-variant (blocks LICM+spill)
    uint32_t zoff = 0;
    asm volatile("" : "+v"(zoff));
    const char* wb = smem + zoff;                 // weight base
    const char* bb = smem + LDS_B_OFF + zoff;     // bias table base

    // ---- A fragments for x_t (gathered) ----
    const long long node = path[(long long)(p0 + m) * LPATH + t];
    const uint16_t* xrow = xbf + (long long)node * IN_DIM;
    v16bf ax[4];
    #pragma unroll
    for (int kc = 0; kc < 4; ++kc) {
      const int k0 = kc * 32 + khalf * 8;
      v8bf lo = *(const v8bf*)(xrow + k0);
      v8bf hi = *(const v8bf*)(xrow + k0 + 16);
      ax[kc] = __builtin_shufflevector(lo, hi, 0,1,2,3,4,5,6,7,8,9,10,11,12,13,14,15);
    }
    // ---- A fragments for h_{t-1} (from LDS staging, padded rows) ----
    v16bf ah[4];
    #pragma unroll
    for (int kc = 0; kc < 4; ++kc) {
      const int k0 = kc * 32 + khalf * 8;
      const char* hp = hbuf + (size_t)m * H_ROW_B + (size_t)k0 * 2;
      v8bf lo = *(const v8bf*)(hp);
      v8bf hi = *(const v8bf*)(hp + 32);
      ah[kc] = __builtin_shufflevector(lo, hi, 0,1,2,3,4,5,6,7,8,9,10,11,12,13,14,15);
    }

    // ---- per 16-wide output-dim tile: 6 zero-init GEMM tiles + GRU elementwise ----
    #pragma unroll
    for (int dt = 0; dt < 8; ++dt) {
      v8f acc[6];   // 0..2: i_r,i_z,i_n   3..5: h_r,h_z,h_n  (biases folded in later)
      #pragma unroll
      for (int g = 0; g < 6; ++g) {
        v8f z = {0.f,0.f,0.f,0.f,0.f,0.f,0.f,0.f};
        acc[g] = z;
      }
      #pragma unroll
      for (int kc = 0; kc < 4; ++kc) {
        const int krow = kc * 32 + khalf * 8;    // K range this lane carries for B
        #pragma unroll
        for (int g = 0; g < 3; ++g) {
          const size_t rowoff = (size_t)(g * OUT_DIM + dt * 16 + col) * W_ROW_B + (size_t)krow * 2;
          {
            const char* wp = wb + rowoff;              // w_ih
            v8bf blo = *(const v8bf*)(wp);
            v8bf bhi = *(const v8bf*)(wp + 32);
            v16bf bfrag = __builtin_shufflevector(blo, bhi, 0,1,2,3,4,5,6,7,8,9,10,11,12,13,14,15);
            acc[g] = __builtin_amdgcn_wmma_f32_16x16x32_bf16(
                false, ax[kc], false, bfrag, (short)0, acc[g], false, false);
          }
          {
            const char* wp = wb + LDS_W1 + rowoff;     // w_hh
            v8bf blo = *(const v8bf*)(wp);
            v8bf bhi = *(const v8bf*)(wp + 32);
            v16bf bfrag = __builtin_shufflevector(blo, bhi, 0,1,2,3,4,5,6,7,8,9,10,11,12,13,14,15);
            acc[3 + g] = __builtin_amdgcn_wmma_f32_16x16x32_bf16(
                false, ah[kc], false, bfrag, (short)0, acc[3 + g], false, false);
          }
        }
      }
      // bias quad for this lane's column: {b_r, b_z, b_in, b_hn}
      const float4 bv = *(const float4*)(bb + (size_t)(dt * 16 + col) * 16);
      // GRU update (C/D layout: element e -> path row e + 8*khalf, column dt*16+col)
      v8f hn;
      #pragma unroll
      for (int e = 0; e < 8; ++e) {
        const float r  = sigm(acc[0][e] + acc[3][e] + bv.x);
        const float z  = sigm(acc[1][e] + acc[4][e] + bv.y);
        const float nc = tanhf(acc[2][e] + bv.z + r * (acc[5][e] + bv.w));
        hn[e] = (1.0f - z) * nc + z * h[dt][e];
      }
      h[dt] = hn;
      // stage back to LDS (bf16) for next step's recurrent A fragment
      #pragma unroll
      for (int e = 0; e < 8; ++e) {
        const int mm = e + (khalf ? 8 : 0);
        *(uint16_t*)(hbuf + (size_t)mm * H_ROW_B + (size_t)(dt * 16 + col) * 2) = f2bf(hn[e]);
      }
    }
  }

  // ---- store path embedding (f32) ----
  #pragma unroll
  for (int dt = 0; dt < 8; ++dt) {
    #pragma unroll
    for (int e = 0; e < 8; ++e) {
      const int mm = e + (khalf ? 8 : 0);
      path_emb[(long long)(p0 + mm) * OUT_DIM + dt * 16 + col] = h[dt][e];
    }
  }

  // ---- attention logits: e = h . a[:,head]; att_un = exp(leaky_relu(e)) ----
  #pragma unroll
  for (int it = 0; it < 2; ++it) {
    const int head = khalf + 2 * it;
    float e = 0.0f;
    #pragma unroll 8
    for (int d = 0; d < OUT_DIM; ++d)
      e += bf2f(*(const uint16_t*)(hbuf + (size_t)m * H_ROW_B + (size_t)d * 2)) *
           amat[d * HEADS + head];
    const float l   = e > 0.0f ? e : 0.2f * e;
    const float att = __expf(l);
    att_un[(long long)(p0 + m) * HEADS + head] = att;
    const int row = (int)path[(long long)(p0 + m) * LPATH + (LPATH - 1)];
    atomicAdd(att_sum + (long long)row * HEADS + head, att);
  }
}

// ---------------- phase 2: normalize + segment-sum scatter ----------------

__global__ __launch_bounds__(OUT_DIM)
void k_scatter(const long long* __restrict__ path,
               const float* __restrict__ path_emb,
               const float* __restrict__ att_un,
               const float* __restrict__ att_sum,
               float* __restrict__ out)       // N x (HEADS*OUT_DIM)
{
  const int p = blockIdx.x;
  const int d = threadIdx.x;
  const int row = (int)path[(long long)p * LPATH + (LPATH - 1)];
  const float emb = path_emb[(long long)p * OUT_DIM + d];
  #pragma unroll
  for (int hh = 0; hh < HEADS; ++hh) {
    const float w = att_un[(long long)p * HEADS + hh] / att_sum[(long long)row * HEADS + hh];
    atomicAdd(out + (long long)row * (HEADS * OUT_DIM) + hh * OUT_DIM + d, w * emb);
  }
}

// ---------------- launch ----------------

extern "C" void kernel_launch(void* const* d_in, const int* in_sizes, int n_in,
                              void* d_out, int out_size, void* d_ws, size_t ws_size,
                              hipStream_t stream) {
  (void)in_sizes; (void)n_in; (void)out_size; (void)ws_size;

  const float*     x     = (const float*)d_in[0];
  const long long* path  = (const long long*)d_in[1];
  const float*     w_ih  = (const float*)d_in[2];
  const float*     w_hh  = (const float*)d_in[3];
  const float*     b_ih  = (const float*)d_in[4];
  const float*     b_hh  = (const float*)d_in[5];
  const float*     a     = (const float*)d_in[6];
  float*           out   = (float*)d_out;

  // workspace carve-out (256B aligned)
  char*  ws  = (char*)d_ws;
  size_t off = 0;
  auto carve = [&](size_t bytes) -> char* {
    char* p = ws + off;
    off = (off + bytes + 255) & ~(size_t)255;
    return p;
  };
  uint16_t* xbf      = (uint16_t*)carve((size_t)N_NODES * IN_DIM * 2);
  uint16_t* wihbf    = (uint16_t*)carve((size_t)3 * OUT_DIM * IN_DIM * 2);
  uint16_t* whhbf    = (uint16_t*)carve((size_t)3 * OUT_DIM * IN_DIM * 2);
  float*    path_emb = (float*)carve((size_t)P_PATHS * OUT_DIM * 4);
  float*    att_un   = (float*)carve((size_t)P_PATHS * HEADS * 4);
  float*    att_sum  = (float*)carve((size_t)N_NODES * HEADS * 4);

  const int nx = N_NODES * IN_DIM;
  const int nw = 3 * OUT_DIM * IN_DIM;
  k_cvt_bf16<<<(nx + 255) / 256, 256, 0, stream>>>(x, xbf, nx);
  k_cvt_bf16<<<(nw + 255) / 256, 256, 0, stream>>>(w_ih, wihbf, nw);
  k_cvt_bf16<<<(nw + 255) / 256, 256, 0, stream>>>(w_hh, whhbf, nw);
  k_zero<<<2048, 256, 0, stream>>>(out, (long long)N_NODES * HEADS * OUT_DIM);
  k_zero<<<256, 256, 0, stream>>>(att_sum, (long long)N_NODES * HEADS);

  const int tiles = (P_PATHS + 16 * WAVES - 1) / (16 * WAVES);
  k_gru_paths<<<tiles, 32 * WAVES, SMEM_BYTES, stream>>>(xbf, path, wihbf, whhbf,
                                                         b_ih, b_hh, a,
                                                         path_emb, att_un, att_sum);
  k_scatter<<<P_PATHS, OUT_DIM, 0, stream>>>(path, path_emb, att_un, att_sum, out);
}
